// YourGNNModel_53111565582842
// MI455X (gfx1250) — matrix operand order repaired and checked by reading, not compiled, not run on hardware
//
#include <hip/hip_runtime.h>

#define NN 100000
#define NE 3200000

typedef __attribute__((ext_vector_type(16))) __bf16 v16bf;
typedef __attribute__((ext_vector_type(8)))  float  v8f;

union Frag16 { v16bf v; uint4 q[2]; };

// ---------------- utility kernels ----------------

__global__ void zero_kernel(float* __restrict__ p, long long n) {
  long long i = (long long)blockIdx.x * blockDim.x + threadIdx.x;
  long long stride = (long long)gridDim.x * blockDim.x;
  for (; i < n; i += stride) p[i] = 0.0f;
}

// out[i] = bf16(in[i])
__global__ void cvt_bf16_kernel(const float* __restrict__ in, __bf16* __restrict__ out,
                                long long n) {
  long long i = (long long)blockIdx.x * blockDim.x + threadIdx.x;
  if (i >= n) return;
  out[i] = (__bf16)in[i];
}

// BT[n*K + k] = bf16(B[k*N + n])   (B is KxN row-major fp32)
__global__ void transpose_cvt_kernel(const float* __restrict__ B, __bf16* __restrict__ BT,
                                     int K, int N) {
  int i = blockIdx.x * blockDim.x + threadIdx.x;
  if (i >= K * N) return;
  int k = i / N, n = i % N;
  BT[(size_t)n * K + k] = (__bf16)B[i];
}

__global__ void degree_kernel(const int* __restrict__ src, const int* __restrict__ dst,
                              float* __restrict__ outdeg, float* __restrict__ indeg,
                              int n_edges) {
  int e = blockIdx.x * blockDim.x + threadIdx.x;
  if (e >= n_edges) return;
  atomicAdd(&outdeg[src[e]], 1.0f);
  atomicAdd(&indeg[dst[e]], 1.0f);
}

__global__ void norm_kernel(const float* __restrict__ outdeg, const float* __restrict__ indeg,
                            float* __restrict__ nrm_out, float* __restrict__ nrm_in, int n) {
  int i = blockIdx.x * blockDim.x + threadIdx.x;
  if (i >= n) return;
  nrm_out[i] = rsqrtf(fmaxf(outdeg[i], 1.0f));
  nrm_in[i]  = rsqrtf(fmaxf(indeg[i], 1.0f));
}

// ---------------- WMMA GEMM: C[m,n] = (sum_k A[m,k]*BT[n,k]) * scale[m] ----------------
// A:  M x K row-major bf16.  BT: N x K row-major bf16 (i.e. B transposed).
// Block = 128 threads = 4 waves; block tile = 16(M) x 64(N); wave w owns n-slice w*16.
// The 16xK A tile is shared by all 4 waves -> staged once in LDS.
// M%16==0, N%64==0, K%32==0: no divergence, EXEC all-1s for WMMA (ISA 7.12 requirement).

template <int N, int K>
__global__ void gemm_wmma_bf16(const __bf16* __restrict__ A, const __bf16* __restrict__ BT,
                               float* __restrict__ C, const float* __restrict__ scale) {
  // A tile: 16 rows x K bf16 = 2*16*K bytes, staged as 16-byte chunks
  constexpr int CHUNKS = 16 * K * 2 / 16;
  __shared__ uint4 shA4[CHUNKS];
  __bf16* shA = (__bf16*)shA4;

  const int tid  = threadIdx.x;
  const int lane = tid & 31;
  const int wave = tid >> 5;
  const int m0 = blockIdx.x * 16;
  const int n0 = blockIdx.y * 64 + wave * 16;

  // Rows m0..m0+15 are contiguous (full K rows) -> linear block copy to LDS.
  const uint4* gA = (const uint4*)(A + (size_t)m0 * K);
#pragma unroll
  for (int c = 0; c < CHUNKS / 128; ++c) shA4[tid + 128 * c] = gA[tid + 128 * c];
  __syncthreads();

  const int hi   = lane >> 4;   // which half-wave
  const int ml   = lane & 15;   // A row / B column within tile
  const int ncol = n0 + ml;

  const __bf16* btcol = BT + (size_t)ncol * K;

  v8f acc = {};
#pragma unroll
  for (int ks = 0; ks < K; ks += 32) {
    // A fragment (16x32): lane half selects K-base 0/8; elems 0..7 = kbA+0..7,
    // elems 8..15 = kbA+16..23  (ISA 7.12.2 16-bit A layout)
    const int kbA = ks + (hi << 3);
    Frag16 a;
    a.q[0] = *(const uint4*)(shA + ml * K + kbA);
    a.q[1] = *(const uint4*)(shA + ml * K + kbA + 16);

    // B fragment (32x16): lane half selects K-base 0/8; elems 0..7 = kbB+0..7, elems 8..15 = kbB+16..23
    const int kbB = ks + (hi << 3);
    Frag16 b;
    b.q[0] = *(const uint4*)(btcol + kbB);
    b.q[1] = *(const uint4*)(btcol + kbB + 16);

    acc = __builtin_amdgcn_wmma_f32_16x16x32_bf16(
        false, a.v, false, b.v, (short)0, acc, false, false);
  }

  // C/D layout: VGPR v -> row = m0 + v + 8*hi, col = ncol
#pragma unroll
  for (int v = 0; v < 8; ++v) {
    const int row = m0 + v + (hi << 3);
    C[(size_t)row * N + ncol] = acc[v] * scale[row];
  }
}

// ---------------- edge-parallel segment-sum (float4 slices) ----------------
// agg[dst[e], :] += msg[src[e], :].  One thread = one float4 slice of one edge row.

template <int D>
__global__ void edge_agg_kernel(const float* __restrict__ msg, const int* __restrict__ src,
                                const int* __restrict__ dst, float* __restrict__ agg,
                                long long total) {  // total = NE * (D/4)
  constexpr int C4 = D / 4;
  long long gid = (long long)blockIdx.x * blockDim.x + threadIdx.x;
  if (gid >= total) return;
  const int e = (int)(gid / C4);
  const int c = (int)(gid % C4);
  const int s = src[e];
  const int d = dst[e];
  const float4 v = *(const float4*)(msg + (size_t)s * D + c * 4);
  float* ap = agg + (size_t)d * D + c * 4;
  atomicAdd(ap + 0, v.x);
  atomicAdd(ap + 1, v.y);
  atomicAdd(ap + 2, v.z);
  atomicAdd(ap + 3, v.w);
}

// ---------------- fused pointwise epilogues ----------------

// hbf = bf16( relu(agg * norm_in[i] + b1[f]) * dropout_mask )   (D=128)
__global__ void epilogue1_kernel(const float* __restrict__ agg, const float* __restrict__ nrm_in,
                                 const float* __restrict__ b1, const float* __restrict__ mask,
                                 __bf16* __restrict__ hbf, long long total) {
  long long gid = (long long)blockIdx.x * blockDim.x + threadIdx.x;
  if (gid >= total) return;
  const int i = (int)(gid >> 7);
  const int f = (int)(gid & 127);
  float v = agg[gid] * nrm_in[i] + b1[f];
  hbf[gid] = (__bf16)(fmaxf(v, 0.0f) * mask[gid]);
}

// out = out * norm_in[i] + b2[f]   (in place, D=64)
__global__ void epilogue2_kernel(float* __restrict__ out, const float* __restrict__ nrm_in,
                                 const float* __restrict__ b2, long long total) {
  long long gid = (long long)blockIdx.x * blockDim.x + threadIdx.x;
  if (gid >= total) return;
  const int i = (int)(gid >> 6);
  const int f = (int)(gid & 63);
  out[gid] = out[gid] * nrm_in[i] + b2[f];
}

// ---------------- launch ----------------

extern "C" void kernel_launch(void* const* d_in, const int* in_sizes, int n_in,
                              void* d_out, int out_size, void* d_ws, size_t ws_size,
                              hipStream_t stream) {
  const float* X    = (const float*)d_in[0];
  const float* W1   = (const float*)d_in[1];
  const float* b1   = (const float*)d_in[2];
  const float* W2   = (const float*)d_in[3];
  const float* b2   = (const float*)d_in[4];
  const float* mask = (const float*)d_in[5];
  const int*   src  = (const int*)d_in[6];
  const int*   dst  = (const int*)d_in[7];
  float* out = (float*)d_out;

  // ---- workspace layout (fp32 region first, then bf16 region; all 16B aligned) ----
  float* ws      = (float*)d_ws;
  float* outdeg  = ws;                          // [NN]
  float* indeg   = ws + (size_t)NN;             // [NN]
  float* nrm_out = ws + (size_t)2 * NN;         // [NN]
  float* nrm_in  = ws + (size_t)3 * NN;         // [NN]
  float* tmp1    = ws + (size_t)4 * NN;         // [NN*128] f32 (GEMM out; reused as tmp2)
  float* agg1    = tmp1 + (size_t)NN * 128;     // [NN*128] f32
  __bf16* Xbf    = (__bf16*)(agg1 + (size_t)NN * 128);  // [NN*128] bf16
  __bf16* hbf    = Xbf + (size_t)NN * 128;               // [NN*128] bf16
  __bf16* W1T    = hbf + (size_t)NN * 128;               // [128*128] bf16
  __bf16* W2T    = W1T + 128 * 128;                      // [64*128]  bf16
  float*  tmp2   = tmp1;                                 // [NN*64] reuse

  // ---- zero accumulators ----
  zero_kernel<<<2048, 256, 0, stream>>>(outdeg, (long long)4 * NN);
  zero_kernel<<<8192, 256, 0, stream>>>(agg1, (long long)NN * 128);
  zero_kernel<<<8192, 256, 0, stream>>>(out, (long long)NN * 64);

  // ---- precision prep: X -> bf16, W1/W2 -> transposed bf16 ----
  {
    long long n = (long long)NN * 128;
    cvt_bf16_kernel<<<(unsigned)((n + 255) / 256), 256, 0, stream>>>(X, Xbf, n);
  }
  transpose_cvt_kernel<<<(128 * 128 + 255) / 256, 256, 0, stream>>>(W1, W1T, 128, 128);
  transpose_cvt_kernel<<<(128 * 64 + 255) / 256, 256, 0, stream>>>(W2, W2T, 128, 64);

  // ---- degrees + symmetric norms ----
  degree_kernel<<<(NE + 255) / 256, 256, 0, stream>>>(src, dst, outdeg, indeg, NE);
  norm_kernel<<<(NN + 255) / 256, 256, 0, stream>>>(outdeg, indeg, nrm_out, nrm_in, NN);

  // ---- layer 1: tmp1 = (X @ W1) * norm_out ----
  {
    dim3 grid(NN / 16, 128 / 64);
    gemm_wmma_bf16<128, 128><<<grid, 128, 0, stream>>>(Xbf, W1T, tmp1, nrm_out);
  }
  // agg1[dst] += tmp1[src]
  {
    long long total = (long long)NE * (128 / 4);
    edge_agg_kernel<128><<<(unsigned)((total + 255) / 256), 256, 0, stream>>>(
        tmp1, src, dst, agg1, total);
  }
  // hbf = bf16(relu(agg1 * norm_in + b1) * mask)
  {
    long long total = (long long)NN * 128;
    epilogue1_kernel<<<(unsigned)((total + 255) / 256), 256, 0, stream>>>(
        agg1, nrm_in, b1, mask, hbf, total);
  }

  // ---- layer 2: tmp2 = (h @ W2) * norm_out ----
  {
    dim3 grid(NN / 16, 64 / 64);
    gemm_wmma_bf16<64, 128><<<grid, 128, 0, stream>>>(hbf, W2T, tmp2, nrm_out);
  }
  // out[dst] += tmp2[src]
  {
    long long total = (long long)NE * (64 / 4);
    edge_agg_kernel<64><<<(unsigned)((total + 255) / 256), 256, 0, stream>>>(
        tmp2, src, dst, out, total);
  }
  // out = out * norm_in + b2 (in place)
  {
    long long total = (long long)NN * 64;
    epilogue2_kernel<<<(unsigned)((total + 255) / 256), 256, 0, stream>>>(
        out, nrm_in, b2, total);
  }
}
